// TrustAwareGAT_75883482185854
// MI455X (gfx1250) — matrix operand order repaired and checked by reading, not compiled, not run on hardware
//
#include <hip/hip_runtime.h>
#include <cstdint>

#define NNODES 50000
#define NEDGES 500000
#define ETOT   (NEDGES + NNODES)
#define INDIM  128
#define HC1    256   // H1*C1
#define HC2    64    // H2*C2

typedef _Float16 half_t;
typedef __attribute__((ext_vector_type(16))) _Float16 v16h;
typedef __attribute__((ext_vector_type(8)))  float    v8f;

static inline int cdiv_ll(long long a, int b) { return (int)((a + b - 1) / b); }

// ---------------------------------------------------------------- utilities

__global__ void fill_u32_kernel(uint32_t* __restrict__ p, uint32_t v, long long n) {
  long long i = (long long)blockIdx.x * blockDim.x + threadIdx.x;
  if (i < n) p[i] = v;
}

// ------------------------------------------------- self-loop attr (fill_value='mean')

__global__ void loop_attr_accum_kernel(const int* __restrict__ dst, const float* __restrict__ ea,
                                       float* __restrict__ sum, float* __restrict__ cnt) {
  int i = blockIdx.x * blockDim.x + threadIdx.x;
  if (i < NEDGES) {
    int d = dst[i];
    atomicAdd(&sum[d], ea[i]);
    atomicAdd(&cnt[d], 1.0f);
  }
}

__global__ void loop_attr_final_kernel(float* __restrict__ sum, const float* __restrict__ cnt) {
  int i = blockIdx.x * blockDim.x + threadIdx.x;
  if (i < NNODES) sum[i] = sum[i] / fmaxf(cnt[i], 1.0f);
}

// ------------------------------------------ WMMA operand swizzles (per-lane contiguous)

// A-matrix 16x32 f16, as packed here: lane = hi*16 + (m&15) holds
// K in [hi*16 .. hi*16+15] of its 32-chunk. Flattened index of
// element (m,k) in the swizzled buffer so each lane's 16 halves are contiguous:
__device__ inline size_t a_swz_index(int m, int k, int K) {
  int mt = m >> 4, l15 = m & 15;
  int kc = k >> 5, r = k & 31;
  int hi = (r >> 4) & 1;                     // which half-wave owns this K
  int j  = r & 15;                           // position within the lane's v16h
  return ((size_t)(mt * (K >> 5) + kc) * 32 + (hi << 4) + l15) * 16 + j;
}

__global__ void cvt_swizzle_a_kernel(const float* __restrict__ A, half_t* __restrict__ Asw,
                                     long long MK, int K) {
  long long t = (long long)blockIdx.x * blockDim.x + threadIdx.x;
  if (t >= MK) return;
  int m = (int)(t / K), k = (int)(t % K);
  Asw[a_swz_index(m, k, K)] = (half_t)A[t];
}

// B-matrix 32x16 f16, as packed here: lane holds column (lane&15), K block (lane>>4)*16..+15.
// Swizzled so each lane's v16h (16 consecutive K of one column) is contiguous.
__global__ void cvt_swizzle_b_kernel(const float* __restrict__ B, half_t* __restrict__ Bsw,
                                     int K, int N) {
  int t = blockIdx.x * blockDim.x + threadIdx.x;
  if (t >= K * N) return;
  int k = t / N, n = t % N;
  size_t idx = ((size_t)((n >> 4) * (K >> 4) + (k >> 4)) * 16 + (n & 15)) * 16 + (k & 15);
  Bsw[idx] = (half_t)B[t];
}

// Fused ELU(x+bias) -> swizzled f16 A for layer-2 GEMM
__global__ void elu_bias_swizzle_kernel(const float* __restrict__ agg,
                                        const float* __restrict__ bias,
                                        half_t* __restrict__ Asw, long long n, int K) {
  long long i = (long long)blockIdx.x * blockDim.x + threadIdx.x;
  if (i >= n) return;
  int m = (int)(i / K), k = (int)(i % K);
  float v = agg[i] + bias[k];
  v = (v > 0.f) ? v : (__expf(v) - 1.f);     // elu, alpha=1
  Asw[a_swz_index(m, k, K)] = (half_t)v;
}

// ---------------------------------------------------------------- WMMA GEMM
// C[M,N] = A[M,K] * B[K,N] with both operands pre-swizzled; one wave per 16x16 tile.
template <int N, int K>
__global__ __launch_bounds__(256) void wmma_gemm_kernel(const half_t* __restrict__ Asw,
                                                        const half_t* __restrict__ Bsw,
                                                        float* __restrict__ C, int M) {
  constexpr int NT = N >> 4;   // N tiles
  constexpr int KC = K >> 5;   // K chunks of 32
  int wave = (int)((blockIdx.x * blockDim.x + threadIdx.x) >> 5);
  int lane = threadIdx.x & 31;
  if (wave >= (M >> 4) * NT) return;
  int mt = wave / NT;
  int nt = wave % NT;
  int hi = lane >> 4, l15 = lane & 15;

  const v16h* ap = (const v16h*)Asw + (size_t)mt * KC * 32 + lane;
  const v16h* bp = (const v16h*)Bsw + (size_t)nt * (K >> 4) * 16 + l15;

  v8f c = {};
#pragma unroll
  for (int kc = 0; kc < KC; ++kc) {
    v16h a = ap[(size_t)kc * 32];
    v16h b = bp[(size_t)(kc * 2 + hi) * 16];
    c = __builtin_amdgcn_wmma_f32_16x16x32_f16(false, a, false, b, (short)0, c,
                                               false, false);
  }
  int row = (mt << 4) + (hi << 3);
  int col = (nt << 4) + l15;
#pragma unroll
  for (int r = 0; r < 8; ++r) C[(size_t)(row + r) * N + col] = c[r];
}

// ------------------------------------------------------------- attention pieces

// kappa[h] = sum_c We[h*C+c] * ae[h*C+c]   (edge-score collapses to scalar/head)
template <int H, int C>
__global__ void kappa_kernel(const float* __restrict__ We, const float* __restrict__ ae,
                             float* __restrict__ kappa) {
  int h = threadIdx.x;
  if (h < H) {
    float s = 0.f;
    for (int c = 0; c < C; ++c) s += We[h * C + c] * ae[h * C + c];
    kappa[h] = s;
  }
}

template <int H, int C>
__global__ void node_scores_kernel(const float* __restrict__ hfeat,
                                   const float* __restrict__ a_src,
                                   const float* __restrict__ a_dst,
                                   float* __restrict__ ssrc, float* __restrict__ sdst) {
  int t = blockIdx.x * blockDim.x + threadIdx.x;
  if (t >= NNODES * H) return;
  int n = t / H, h = t % H;
  const float* hp = hfeat + (size_t)n * H * C + h * C;
  const float* ap = a_src + h * C;
  const float* dp = a_dst + h * C;
  float s1 = 0.f, s2 = 0.f;
  for (int c = 0; c < C; ++c) { float v = hp[c]; s1 += v * ap[c]; s2 += v * dp[c]; }
  ssrc[t] = s1;
  sdst[t] = s2;
}

__device__ inline float atomicMaxF(float* addr, float v) {
  if (v >= 0.f) return __int_as_float(atomicMax((int*)addr, __float_as_int(v)));
  return __uint_as_float(atomicMin((unsigned int*)addr, __float_as_uint(v)));
}

template <int H>
__global__ void edge_logits_kernel(const int* __restrict__ src, const int* __restrict__ dst,
                                   const float* __restrict__ ea,
                                   const float* __restrict__ loop_attr,
                                   const float* __restrict__ ssrc, const float* __restrict__ sdst,
                                   const float* __restrict__ kappa,
                                   float* __restrict__ logit, float* __restrict__ mx) {
  int i = blockIdx.x * blockDim.x + threadIdx.x;
  if (i >= ETOT) return;
  int s, d; float a;
  if (i < NEDGES) { s = src[i]; d = dst[i]; a = ea[i]; }
  else            { s = d = i - NEDGES;     a = loop_attr[s]; }
#pragma unroll
  for (int h = 0; h < H; ++h) {
    float l = ssrc[s * H + h] + sdst[d * H + h] + a * kappa[h];
    l = (l > 0.f) ? l : 0.2f * l;          // leaky_relu(0.2)
    logit[(size_t)i * H + h] = l;
    atomicMaxF(&mx[d * H + h], l);
  }
}

template <int H>
__global__ void edge_exp_kernel(const int* __restrict__ dst, float* __restrict__ logit,
                                const float* __restrict__ mx, float* __restrict__ den) {
  int i = blockIdx.x * blockDim.x + threadIdx.x;
  if (i >= ETOT) return;
  int d = (i < NEDGES) ? dst[i] : (i - NEDGES);
#pragma unroll
  for (int h = 0; h < H; ++h) {
    float a = __expf(logit[(size_t)i * H + h] - mx[d * H + h]);
    logit[(size_t)i * H + h] = a;
    atomicAdd(&den[d * H + h], a);
  }
}

template <int H, int C>
__global__ void edge_aggregate_kernel(const int* __restrict__ src, const int* __restrict__ dst,
                                      const float* __restrict__ hfeat,
                                      const float* __restrict__ alpha,
                                      const float* __restrict__ den,
                                      float* __restrict__ agg) {
  const int HC = H * C;
  long long t = (long long)blockIdx.x * blockDim.x + threadIdx.x;
  if (t >= (long long)ETOT * HC) return;
  int i  = (int)(t / HC);
  int ch = (int)(t % HC);
  int h  = ch / C;
  int s, d;
  if (i < NEDGES) { s = src[i]; d = dst[i]; } else { s = d = i - NEDGES; }
  float w = alpha[(size_t)i * H + h] / (den[d * H + h] + 1e-16f);
  atomicAdd(&agg[(size_t)d * HC + ch], hfeat[(size_t)s * HC + ch] * w);
}

__global__ void final_bias_kernel(const float* __restrict__ agg, const float* __restrict__ bias,
                                  float* __restrict__ out, long long n, int HC) {
  long long i = (long long)blockIdx.x * blockDim.x + threadIdx.x;
  if (i >= n) return;
  out[i] = agg[i] + bias[i % HC];
}

// ---------------------------------------------------------------- launcher

extern "C" void kernel_launch(void* const* d_in, const int* in_sizes, int n_in,
                              void* d_out, int out_size, void* d_ws, size_t ws_size,
                              hipStream_t stream) {
  (void)in_sizes; (void)n_in; (void)out_size; (void)ws_size;
  const float* x    = (const float*)d_in[0];
  const int*   ei   = (const int*)d_in[1];
  const float* ea   = (const float*)d_in[2];
  const float* W1   = (const float*)d_in[3];
  const float* We1  = (const float*)d_in[4];
  const float* as1  = (const float*)d_in[5];
  const float* ad1  = (const float*)d_in[6];
  const float* ae1  = (const float*)d_in[7];
  const float* b1   = (const float*)d_in[8];
  const float* W2   = (const float*)d_in[9];
  const float* We2  = (const float*)d_in[10];
  const float* as2  = (const float*)d_in[11];
  const float* ad2  = (const float*)d_in[12];
  const float* ae2  = (const float*)d_in[13];
  const float* b2   = (const float*)d_in[14];
  const int* src = ei;
  const int* dst = ei + NEDGES;
  float* out = (float*)d_out;

  // ---- workspace carve-out (256B aligned slices)
  char* ws = (char*)d_ws;
  size_t off = 0;
  auto take = [&](size_t bytes) -> char* {
    char* p = ws + off;
    off = (off + bytes + 255) & ~(size_t)255;
    return p;
  };
  float*  loopattr = (float*)take((size_t)NNODES * 4);
  float*  cnt      = (float*)take((size_t)NNODES * 4);
  float*  kappa1   = (float*)take(4 * 4);
  float*  kappa2   = (float*)take(4);
  half_t* xsw      = (half_t*)take((size_t)NNODES * INDIM * 2);
  half_t* W1sw     = (half_t*)take((size_t)INDIM * HC1 * 2);
  half_t* W2sw     = (half_t*)take((size_t)HC1 * HC2 * 2);
  float*  h1       = (float*)take((size_t)NNODES * HC1 * 4);
  float*  agg1     = (float*)take((size_t)NNODES * HC1 * 4);
  half_t* h1sw     = (half_t*)take((size_t)NNODES * HC1 * 2);
  float*  ssrc     = (float*)take((size_t)NNODES * 4 * 4);
  float*  sdst     = (float*)take((size_t)NNODES * 4 * 4);
  float*  mx       = (float*)take((size_t)NNODES * 4 * 4);
  float*  den      = (float*)take((size_t)NNODES * 4 * 4);
  float*  logit    = (float*)take((size_t)ETOT * 4 * 4);
  // layer-1 h1 region is dead after layer-1 aggregation -> reuse for layer 2
  float* h2   = h1;
  float* agg2 = h1 + (size_t)NNODES * HC2;

  const int T = 256;
  const uint32_t NEG_INF = 0xFF800000u;

  // ---- self-loop edge attr = mean of incoming
  fill_u32_kernel<<<cdiv_ll(NNODES, T), T, 0, stream>>>((uint32_t*)loopattr, 0u, NNODES);
  fill_u32_kernel<<<cdiv_ll(NNODES, T), T, 0, stream>>>((uint32_t*)cnt, 0u, NNODES);
  loop_attr_accum_kernel<<<cdiv_ll(NEDGES, T), T, 0, stream>>>(dst, ea, loopattr, cnt);
  loop_attr_final_kernel<<<cdiv_ll(NNODES, T), T, 0, stream>>>(loopattr, cnt);

  // ---- f16 conversions + WMMA fragment swizzles
  cvt_swizzle_a_kernel<<<cdiv_ll((long long)NNODES * INDIM, T), T, 0, stream>>>(
      x, xsw, (long long)NNODES * INDIM, INDIM);
  cvt_swizzle_b_kernel<<<cdiv_ll((long long)INDIM * HC1, T), T, 0, stream>>>(W1, W1sw, INDIM, HC1);
  cvt_swizzle_b_kernel<<<cdiv_ll((long long)HC1 * HC2, T), T, 0, stream>>>(W2, W2sw, HC1, HC2);

  // ---- layer 1: h1 = x @ W1  (3125*16 tiles, one wave each)
  {
    long long waves = (long long)(NNODES / 16) * (HC1 / 16);
    wmma_gemm_kernel<HC1, INDIM><<<cdiv_ll(waves * 32, T), T, 0, stream>>>(xsw, W1sw, h1, NNODES);
  }
  kappa_kernel<4, 64><<<1, 32, 0, stream>>>(We1, ae1, kappa1);
  node_scores_kernel<4, 64><<<cdiv_ll((long long)NNODES * 4, T), T, 0, stream>>>(h1, as1, ad1, ssrc, sdst);

  fill_u32_kernel<<<cdiv_ll((long long)NNODES * 4, T), T, 0, stream>>>((uint32_t*)mx, NEG_INF, (long long)NNODES * 4);
  fill_u32_kernel<<<cdiv_ll((long long)NNODES * 4, T), T, 0, stream>>>((uint32_t*)den, 0u, (long long)NNODES * 4);
  fill_u32_kernel<<<cdiv_ll((long long)NNODES * HC1, T), T, 0, stream>>>((uint32_t*)agg1, 0u, (long long)NNODES * HC1);

  edge_logits_kernel<4><<<cdiv_ll(ETOT, T), T, 0, stream>>>(src, dst, ea, loopattr, ssrc, sdst, kappa1, logit, mx);
  edge_exp_kernel<4><<<cdiv_ll(ETOT, T), T, 0, stream>>>(dst, logit, mx, den);
  edge_aggregate_kernel<4, 64><<<cdiv_ll((long long)ETOT * HC1, T), T, 0, stream>>>(src, dst, h1, logit, den, agg1);

  // ---- ELU(out1 + b1) -> swizzled f16 A of layer 2 (fused)
  elu_bias_swizzle_kernel<<<cdiv_ll((long long)NNODES * HC1, T), T, 0, stream>>>(
      agg1, b1, h1sw, (long long)NNODES * HC1, HC1);

  // ---- layer 2: h2 = elu_out @ W2
  {
    long long waves = (long long)(NNODES / 16) * (HC2 / 16);
    wmma_gemm_kernel<HC2, HC1><<<cdiv_ll(waves * 32, T), T, 0, stream>>>(h1sw, W2sw, h2, NNODES);
  }
  kappa_kernel<1, 64><<<1, 32, 0, stream>>>(We2, ae2, kappa2);
  node_scores_kernel<1, 64><<<cdiv_ll(NNODES, T), T, 0, stream>>>(h2, as2, ad2, ssrc, sdst);

  fill_u32_kernel<<<cdiv_ll(NNODES, T), T, 0, stream>>>((uint32_t*)mx, NEG_INF, NNODES);
  fill_u32_kernel<<<cdiv_ll(NNODES, T), T, 0, stream>>>((uint32_t*)den, 0u, NNODES);
  fill_u32_kernel<<<cdiv_ll((long long)NNODES * HC2, T), T, 0, stream>>>((uint32_t*)agg2, 0u, (long long)NNODES * HC2);

  edge_logits_kernel<1><<<cdiv_ll(ETOT, T), T, 0, stream>>>(src, dst, ea, loopattr, ssrc, sdst, kappa2, logit, mx);
  edge_exp_kernel<1><<<cdiv_ll(ETOT, T), T, 0, stream>>>(dst, logit, mx, den);
  edge_aggregate_kernel<1, 64><<<cdiv_ll((long long)ETOT * HC2, T), T, 0, stream>>>(src, dst, h2, logit, den, agg2);

  final_bias_kernel<<<cdiv_ll((long long)NNODES * HC2, T), T, 0, stream>>>(agg2, b2, out, (long long)NNODES * HC2, HC2);
}
